// SelfAttention_25340307046459
// MI455X (gfx1250) — matrix operand order repaired and checked by reading, not compiled, not run on hardware
//
#include <hip/hip_runtime.h>
#include <hip/hip_bf16.h>
#include <stdint.h>

typedef _Float16 v16h __attribute__((ext_vector_type(16)));
typedef float    v8f  __attribute__((ext_vector_type(8)));
typedef __fp16   fp16x2 __attribute__((ext_vector_type(2)));

union FragH {
    uint32_t u[8];
    v16h     h;
};

static __device__ __forceinline__ uint32_t pack2h(float a, float b) {
    union { fp16x2 h; uint32_t u; } p;
    p.h = __builtin_amdgcn_cvt_pkrtz(a, b);   // v_cvt_pk_rtz_f16_f32
    return p.u;
}

// ---------------------------------------------------------------------------
// Tile geometry: 128x64 block tile, 4 waves (128 threads), K-step 32.
// Each wave owns a 32x64 strip: 2 A-fragments x 4 B-fragments = 8 WMMAs/step.
// LDS A tile: 128 rows x 32 f16, row stride 40 f16 = 20 u32 (80 B, 16B-mult)
// LDS B tile:  32 rows x 64 f16, row stride 72 f16 = 36 u32 (144 B, 16B-mult)
// Double-buffered: one barrier per K iteration.
// ---------------------------------------------------------------------------
#define LDA_U32 20
#define LDB_U32 36
#define A_TILE_U32 (128 * LDA_U32)
#define B_TILE_U32 (32 * LDB_U32)

static __device__ __forceinline__ void mma_step(const uint32_t* __restrict__ ldsA,
                                                const uint32_t* __restrict__ ldsB,
                                                int wave, int lane, v8f acc[2][4]) {
    FragH a[2];
    const int half = lane >> 4;
#pragma unroll
    for (int p = 0; p < 2; ++p) {
        const int m = wave * 32 + p * 16 + (lane & 15);
#pragma unroll
        for (int j = 0; j < 8; ++j) {
            const int kk = ((j & 4) << 2) + (half << 3) + ((j & 3) << 1);
            a[p].u[j] = ldsA[m * LDA_U32 + (kk >> 1)];
        }
    }
#pragma unroll
    for (int g = 0; g < 4; ++g) {
        FragH b;
#pragma unroll
        for (int s = 0; s < 16; ++s) b.h[s] = ((const _Float16*)ldsB)[(((s & 8) << 1) + (half << 3) + (s & 7)) * (2 * LDB_U32) + g * 16 + (lane & 15)];
#pragma unroll
        for (int p = 0; p < 2; ++p) {
            acc[p][g] = __builtin_amdgcn_wmma_f32_16x16x32_f16(
                false, a[p].h, false, b.h, (short)0, acc[p][g], false, false);
        }
    }
}

// ---------------------------------------------------------------------------
// Kernel 1: y = x @ W + b    (x fp32 [8192,1024], W fp32 [1024,1024])
// Output f16. TRANSPOSED==0 -> out[r*1024 + c]            (Q, V)
// TRANSPOSED==1 -> out[(batch*1024 + c)*2048 + t]          (K stored as Kt)
// ---------------------------------------------------------------------------
template <int TRANSPOSED>
__global__ __launch_bounds__(128, 1) void qkv_gemm(const float* __restrict__ x,
                                                   const float* __restrict__ W,
                                                   const float* __restrict__ bias,
                                                   _Float16* __restrict__ out) {
    __shared__ uint32_t ldsA[2][A_TILE_U32];
    __shared__ uint32_t ldsB[2][B_TILE_U32];
    const int tid = threadIdx.x;
    const int wave = tid >> 5;
    const int lane = tid & 31;
    const int r0 = blockIdx.x * 128;
    const int c0 = blockIdx.y * 64;

    v8f acc[2][4] = {};

    // A tile: 128x32 f32 = 1024 float4, 8 per thread.
    const float4* pA[8];
    uint32_t offA[8];
#pragma unroll
    for (int i = 0; i < 8; ++i) {
        const int idx = tid + 128 * i;
        const int row = idx >> 3;        // 8 float4 per row
        const int pos = idx & 7;         // float4 within row
        pA[i] = (const float4*)(x + (size_t)(r0 + row) * 1024) + pos;
        offA[i] = row * LDA_U32 + pos * 2;
    }
    // B tile: 32x64 f32 = 512 float4, 4 per thread.
    const float4* pB[4];
    uint32_t offB[4];
#pragma unroll
    for (int i = 0; i < 4; ++i) {
        const int idx = tid + 128 * i;
        const int row = idx >> 4;        // 16 float4 per row
        const int pos = idx & 15;
        pB[i] = (const float4*)(W + (size_t)row * 1024 + c0) + pos;
        offB[i] = row * LDB_U32 + pos * 2;
    }

    auto loadA = [&](int buf) {
#pragma unroll
        for (int i = 0; i < 8; ++i) {
            const float4 f = *pA[i];
            uint2 d;
            d.x = pack2h(f.x, f.y);
            d.y = pack2h(f.z, f.w);
            *(uint2*)&ldsA[buf][offA[i]] = d;
        }
    };
    auto loadB = [&](int buf) {
#pragma unroll
        for (int i = 0; i < 4; ++i) {
            const float4 f = *pB[i];
            uint2 d;
            d.x = pack2h(f.x, f.y);
            d.y = pack2h(f.z, f.w);
            *(uint2*)&ldsB[buf][offB[i]] = d;
        }
    };
    auto advance = [&]() {
#pragma unroll
        for (int i = 0; i < 8; ++i) pA[i] += 8;        // +32 floats
#pragma unroll
        for (int i = 0; i < 4; ++i) pB[i] += 8192;     // +32 rows of W
    };

    loadA(0);
    loadB(0);
    advance();
    __syncthreads();

    int buf = 0;
    const int NIT = 1024 / 32;
    for (int it = 0; it < NIT; ++it) {
        if (it + 1 < NIT) {
            loadA(buf ^ 1);
            loadB(buf ^ 1);
            advance();
            if (it + 2 < NIT) {  // L2 prefetch one more tile ahead
                __builtin_prefetch((const void*)pA[0], 0, 1);
                __builtin_prefetch((const void*)pB[0], 0, 1);
            }
        }
        mma_step(ldsA[buf], ldsB[buf], wave, lane, acc);
        __syncthreads();
        buf ^= 1;
    }

    // Epilogue: bias add, f16 store
#pragma unroll
    for (int g = 0; g < 4; ++g) {
        const int c = c0 + g * 16 + (lane & 15);
        const float bv = bias[c];
#pragma unroll
        for (int p = 0; p < 2; ++p) {
#pragma unroll
            for (int i = 0; i < 8; ++i) {
                const int r = r0 + wave * 32 + p * 16 + ((lane >> 4) << 3) + i;
                const _Float16 h = (_Float16)(acc[p][g][i] + bv);
                if (TRANSPOSED == 0) {
                    out[(size_t)r * 1024 + c] = h;
                } else {
                    const int bidx = r >> 11;
                    const int t = r & 2047;
                    out[((size_t)bidx * 1024 + c) * 2048 + t] = h;
                }
            }
        }
    }
}

// ---------------------------------------------------------------------------
// Generic f16 GEMM body: C[128 x 64] tile, A row-major, B row-major,
// K-loop with NIT iterations, b128 global->LDS tile copies.
// ---------------------------------------------------------------------------
struct GemmPtrs {
    const uint4* pA[4];
    uint32_t     offA[4];
    const uint4* pB[2];
    uint32_t     offB[2];
    int          stepA;   // uint4 advance per K-step for A
    int          stepB;   // uint4 advance per K-step for B
};

static __device__ __forceinline__ void f16_gemm_loop(GemmPtrs& gp, int NIT, int wave,
                                                     int lane, uint32_t (*ldsA)[A_TILE_U32],
                                                     uint32_t (*ldsB)[B_TILE_U32],
                                                     v8f acc[2][4]) {
    auto loadA = [&](int buf) {
#pragma unroll
        for (int i = 0; i < 4; ++i) {
            *(uint4*)&ldsA[buf][gp.offA[i]] = *gp.pA[i];
        }
    };
    auto loadB = [&](int buf) {
#pragma unroll
        for (int i = 0; i < 2; ++i) {
            *(uint4*)&ldsB[buf][gp.offB[i]] = *gp.pB[i];
        }
    };
    auto advance = [&]() {
#pragma unroll
        for (int i = 0; i < 4; ++i) gp.pA[i] += gp.stepA;
#pragma unroll
        for (int i = 0; i < 2; ++i) gp.pB[i] += gp.stepB;
    };

    loadA(0);
    loadB(0);
    advance();
    __syncthreads();

    int buf = 0;
    for (int it = 0; it < NIT; ++it) {
        if (it + 1 < NIT) {
            loadA(buf ^ 1);
            loadB(buf ^ 1);
            advance();
            if (it + 2 < NIT) {
                __builtin_prefetch((const void*)gp.pA[0], 0, 1);
                __builtin_prefetch((const void*)gp.pB[0], 0, 1);
            }
        }
        mma_step(ldsA[buf], ldsB[buf], wave, lane, acc);
        __syncthreads();
        buf ^= 1;
    }
}

// ---------------------------------------------------------------------------
// Kernel 2: S[b,q,k] = sum_e Q[b,q,e] * K[b,k,e]
// A = Q f16 [b*2048+q, 1024] row-major, B = Kt f16 [b,1024,2048]
// ---------------------------------------------------------------------------
__global__ __launch_bounds__(128, 1) void scores_gemm(const _Float16* __restrict__ Q,
                                                      const _Float16* __restrict__ Kt,
                                                      float* __restrict__ S) {
    __shared__ uint32_t ldsA[2][A_TILE_U32];
    __shared__ uint32_t ldsB[2][B_TILE_U32];
    const int tid = threadIdx.x;
    const int wave = tid >> 5;
    const int lane = tid & 31;
    const int b = blockIdx.z;
    const int q0 = blockIdx.x * 128;
    const int n0 = blockIdx.y * 64;
    const uint32_t* Qu = (const uint32_t*)Q;
    const uint32_t* Ku = (const uint32_t*)Kt;

    v8f acc[2][4] = {};
    GemmPtrs gp;
    // A tile: 128 rows x 16 u32 = 512 uint4, 4 per thread.
#pragma unroll
    for (int i = 0; i < 4; ++i) {
        const int idx = tid + 128 * i;
        const int row = idx >> 2;         // 4 uint4 per row
        const int qp = idx & 3;
        gp.pA[i] = (const uint4*)(Qu + (size_t)(b * 2048 + q0 + row) * 512) + qp;
        gp.offA[i] = row * LDA_U32 + qp * 4;
    }
    // B tile: 32 rows x 32 u32 = 256 uint4, 2 per thread.
#pragma unroll
    for (int i = 0; i < 2; ++i) {
        const int idx = tid + 128 * i;
        const int row = idx >> 3;         // 8 uint4 per row
        const int qp = idx & 7;
        gp.pB[i] = (const uint4*)(Ku + ((size_t)b * 1024 + row) * 1024 + (n0 >> 1)) + qp;
        gp.offB[i] = row * LDB_U32 + qp * 4;
    }
    gp.stepA = 4;      // +32 f16 along e
    gp.stepB = 8192;   // +32 rows of Kt (32 * 1024 u32)

    f16_gemm_loop(gp, 1024 / 32, wave, lane, ldsA, ldsB, acc);

#pragma unroll
    for (int g = 0; g < 4; ++g) {
        const int n = n0 + g * 16 + (lane & 15);
#pragma unroll
        for (int p = 0; p < 2; ++p) {
#pragma unroll
            for (int i = 0; i < 8; ++i) {
                const int q = q0 + wave * 32 + p * 16 + ((lane >> 4) << 3) + i;
                S[((size_t)b * 2048 + q) * 2048 + n] = acc[p][g][i];
            }
        }
    }
}

// ---------------------------------------------------------------------------
// Kernel 3: row softmax over 2048 elements; fp32 in, f16 alpha out.
// ---------------------------------------------------------------------------
__global__ void softmax_rows(const float* __restrict__ S,
                             _Float16* __restrict__ alpha) {
    __shared__ float red[256];
    const int row = blockIdx.x;
    const int tid = threadIdx.x;
    const float* srow = S + (size_t)row * 2048;

    float v[8];
    float m = -3.402823466e38f;
#pragma unroll
    for (int i = 0; i < 8; ++i) {
        v[i] = srow[tid + 256 * i];
        m = fmaxf(m, v[i]);
    }
    red[tid] = m;
    __syncthreads();
    for (int s = 128; s > 0; s >>= 1) {
        if (tid < s) red[tid] = fmaxf(red[tid], red[tid + s]);
        __syncthreads();
    }
    const float M = red[0];
    __syncthreads();

    float e[8];
    float sum = 0.f;
#pragma unroll
    for (int i = 0; i < 8; ++i) {
        e[i] = __expf(v[i] - M);
        sum += e[i];
    }
    red[tid] = sum;
    __syncthreads();
    for (int s = 128; s > 0; s >>= 1) {
        if (tid < s) red[tid] += red[tid + s];
        __syncthreads();
    }
    const float inv = 1.f / red[0];

    _Float16* arow = alpha + (size_t)row * 2048;
#pragma unroll
    for (int i = 0; i < 8; ++i) {
        arow[tid + 256 * i] = (_Float16)(e[i] * inv);
    }
}

// ---------------------------------------------------------------------------
// Kernel 4: O[b,q,e] = sum_k alpha[b,q,k] * V[b,k,e]   (fp32 out)
// ---------------------------------------------------------------------------
__global__ __launch_bounds__(128, 1) void out_gemm(const _Float16* __restrict__ alpha,
                                                   const _Float16* __restrict__ V,
                                                   float* __restrict__ O) {
    __shared__ uint32_t ldsA[2][A_TILE_U32];
    __shared__ uint32_t ldsB[2][B_TILE_U32];
    const int tid = threadIdx.x;
    const int wave = tid >> 5;
    const int lane = tid & 31;
    const int b = blockIdx.z;
    const int q0 = blockIdx.x * 128;
    const int e0 = blockIdx.y * 64;
    const uint32_t* Au = (const uint32_t*)alpha;
    const uint32_t* Vu = (const uint32_t*)V;

    v8f acc[2][4] = {};
    GemmPtrs gp;
    // A tile: alpha rows (stride 1024 u32), 128 rows x 16 u32.
#pragma unroll
    for (int i = 0; i < 4; ++i) {
        const int idx = tid + 128 * i;
        const int row = idx >> 2;
        const int qp = idx & 3;
        gp.pA[i] = (const uint4*)(Au + (size_t)(b * 2048 + q0 + row) * 1024) + qp;
        gp.offA[i] = row * LDA_U32 + qp * 4;
    }
    // B tile: V rows (stride 512 u32), 32 rows x 32 u32.
#pragma unroll
    for (int i = 0; i < 2; ++i) {
        const int idx = tid + 128 * i;
        const int row = idx >> 3;
        const int qp = idx & 7;
        gp.pB[i] = (const uint4*)(Vu + (size_t)(b * 2048 + row) * 512 + (e0 >> 1)) + qp;
        gp.offB[i] = row * LDB_U32 + qp * 4;
    }
    gp.stepA = 4;      // +32 f16 along k
    gp.stepB = 4096;   // +32 rows of V (32 * 512 u32)

    f16_gemm_loop(gp, 2048 / 32, wave, lane, ldsA, ldsB, acc);

#pragma unroll
    for (int g = 0; g < 4; ++g) {
        const int e = e0 + g * 16 + (lane & 15);
#pragma unroll
        for (int p = 0; p < 2; ++p) {
#pragma unroll
            for (int i = 0; i < 8; ++i) {
                const int q = q0 + wave * 32 + p * 16 + ((lane >> 4) << 3) + i;
                O[((size_t)b * 2048 + q) * 1024 + e] = acc[p][g][i];
            }
        }
    }
}

// ---------------------------------------------------------------------------
// Workspace layout (bytes):
//   [0,   16M)  Q   f16 [8192,1024]
//   [16M, 32M)  Kt  f16 [4,1024,2048]
//   [32M, 48M)  V   f16 [8192,1024]
//   [48M,112M)  S   f32 [4,2048,2048]
// alpha f16 [4,2048,2048] = 32M overlays [0,32M) after Q/Kt are dead.
// ---------------------------------------------------------------------------
extern "C" void kernel_launch(void* const* d_in, const int* in_sizes, int n_in,
                              void* d_out, int out_size, void* d_ws, size_t ws_size,
                              hipStream_t stream) {
    (void)in_sizes; (void)n_in; (void)out_size; (void)ws_size;

    const float* x  = (const float*)d_in[0];
    const float* Wq = (const float*)d_in[1];
    const float* bq = (const float*)d_in[2];
    const float* Wk = (const float*)d_in[3];
    const float* bk = (const float*)d_in[4];
    const float* Wv = (const float*)d_in[5];
    const float* bv = (const float*)d_in[6];

    char* ws = (char*)d_ws;
    _Float16* Q     = (_Float16*)(ws);
    _Float16* Kt    = (_Float16*)(ws + ((size_t)16 << 20));
    _Float16* V     = (_Float16*)(ws + ((size_t)32 << 20));
    float*    S     = (float*)   (ws + ((size_t)48 << 20));
    _Float16* alpha = (_Float16*)(ws);  // overlays Q+Kt (dead after scores)
    float*    O     = (float*)d_out;

    const dim3 blk(128);
    const dim3 g_qkv(64, 16, 1);  // 8192/128 row tiles x 1024/64 col tiles

    qkv_gemm<0><<<g_qkv, blk, 0, stream>>>(x, Wq, bq, Q);
    qkv_gemm<1><<<g_qkv, blk, 0, stream>>>(x, Wk, bk, Kt);
    qkv_gemm<0><<<g_qkv, blk, 0, stream>>>(x, Wv, bv, V);

    scores_gemm<<<dim3(16, 32, 4), blk, 0, stream>>>(Q, Kt, S);

    softmax_rows<<<dim3(8192), dim3(256), 0, stream>>>(S, alpha);

    out_gemm<<<dim3(16, 16, 4), blk, 0, stream>>>(alpha, V, O);
}
